// TokenPairModeling_66803921322302
// MI455X (gfx1250) — matrix operand order repaired and checked by reading, not compiled, not run on hardware
//
#include <hip/hip_runtime.h>

// ---------------------------------------------------------------------------
// TokenPairModeling for MI455X (gfx1250): everything routed through a single
// bf16-WMMA tiled GEMM (v_wmma_f32_16x16x32_bf16, fp32 accumulate).
// Block 128x128x32, 8 waves, 64x32 wave tile (8 WMMAs / K-step / wave),
// double-buffered LDS, one barrier per K-step, global->LDS store at the top
// of each iteration so the staging loads get a full iteration of latency
// hiding behind the WMMAs.
// ---------------------------------------------------------------------------

#define BB  16
#define SS  384
#define HH  768
#define DKK 64
#define PP  73920L   // S*(S+1)/2

typedef __bf16 bf16_t;
typedef __bf16 v16bf __attribute__((ext_vector_type(16)));
typedef __bf16 v8bf  __attribute__((ext_vector_type(8)));
typedef float  v8f   __attribute__((ext_vector_type(8)));

// ---------------- generic batched GEMM:  C[z] = alpha * A[z] * B[z]^T (+bias)
// A: [M,K] row-major, B: [N,K] row-major (i.e. weight layout [out,in])
struct GemmArgs {
  const bf16_t* A; const bf16_t* B; const float* bias; void* C;
  long sA, sB, sC;          // batch strides (elements)
  int  aDiv, bMod;          // A batch index = z/aDiv, B batch index = z%bMod
  int  M, N, K, lda, ldb, ldc;
  float alpha; int flags;   // 1=relu, 2=transposed store C[n,m], 4=bf16 output
};

#define TBM 128
#define TBN 128
#define TBK 32
#define LDS_STR 40   // padded LDS row stride (bf16 elems)

__global__ __launch_bounds__(256) void gemm_bf16_wmma(GemmArgs g) {
  __shared__ bf16_t As[2][TBM * LDS_STR];
  __shared__ bf16_t Bs[2][TBN * LDS_STR];
  const int z  = blockIdx.z;
  const bf16_t* A  = g.A + g.sA * (long)(z / g.aDiv);
  const bf16_t* Bw = g.B + g.sB * (long)(z % g.bMod);
  const int m0 = blockIdx.y * TBM;
  const int n0 = blockIdx.x * TBN;
  const int tid  = threadIdx.x;
  const int lane = tid & 31;
  const int wid  = tid >> 5;
  const int wm = wid & 1;        // 0..1 -> 64-row strip
  const int wn = wid >> 1;       // 0..3 -> 32-col strip
  v8f acc[4][2] = {};

  // WMMA fragment addressing (contiguous k order: lane half h holds k = 16h..16h+15 of A and of B)
  const int fm  = lane & 15;
  const int k0a = (lane >> 4) * 16;   // A: chunk  [k0a..k0a+15]
  const int k0b = (lane >> 4) * 16;   // B: chunk  [k0b..k0b+15]

  // global staging coords: 256 threads move 128x32 of A and of B per step
  const int srow = (tid * 16) >> 5, scol = (tid * 16) & 31;   // 16 elems/thread
  int gbn = n0 + srow; if (gbn >= g.N) gbn = g.N - 1;         // clamp (guarded)
  const bf16_t* ag = A  + (long)(m0 + srow) * g.lda + scol;
  const bf16_t* bg = Bw + (long)gbn * g.ldb + scol;
  const int sidx = srow * LDS_STR + scol;

  const int nK = g.K / TBK;
  // ---- prologue: stage K-slab 0 directly into buffer 0, preload slab 1 regs
  {
    v8bf a0 = *(const v8bf*)(ag);
    v8bf a1 = *(const v8bf*)(ag + 8);
    v8bf b0 = *(const v8bf*)(bg);
    v8bf b1 = *(const v8bf*)(bg + 8);
    ag += TBK; bg += TBK;
    *(v8bf*)&As[0][sidx]     = a0;
    *(v8bf*)&As[0][sidx + 8] = a1;
    *(v8bf*)&Bs[0][sidx]     = b0;
    *(v8bf*)&Bs[0][sidx + 8] = b1;
  }
  v8bf na0 = {}, na1 = {}, nb0 = {}, nb1 = {};
  if (nK > 1) {
    na0 = *(const v8bf*)(ag);
    na1 = *(const v8bf*)(ag + 8);
    nb0 = *(const v8bf*)(bg);
    nb1 = *(const v8bf*)(bg + 8);
    __builtin_prefetch(ag + TBK, 0, 3);
    __builtin_prefetch(bg + TBK, 0, 3);
    ag += TBK; bg += TBK;
  }

  for (int kk = 0; kk < nK; ++kk) {
    const bf16_t* curA = As[kk & 1];
    const bf16_t* curB = Bs[kk & 1];
    __syncthreads();   // buffer (kk&1) fully written; other buffer free

    // top-of-iteration: commit staged regs (slab kk+1) to the idle buffer,
    // then immediately issue the loads for slab kk+2 -> a full iteration of
    // WMMA work hides their latency.
    if (kk + 1 < nK) {
      bf16_t* nA = As[(kk + 1) & 1];
      bf16_t* nB = Bs[(kk + 1) & 1];
      *(v8bf*)&nA[sidx]     = na0;
      *(v8bf*)&nA[sidx + 8] = na1;
      *(v8bf*)&nB[sidx]     = nb0;
      *(v8bf*)&nB[sidx + 8] = nb1;
      if (kk + 2 < nK) {
        na0 = *(const v8bf*)(ag);
        na1 = *(const v8bf*)(ag + 8);
        nb0 = *(const v8bf*)(bg);
        nb1 = *(const v8bf*)(bg + 8);
        __builtin_prefetch(ag + TBK, 0, 3);   // global_prefetch_b8
        __builtin_prefetch(bg + TBK, 0, 3);
        ag += TBK; bg += TBK;
      }
    }

    // B fragments (reused across 4 M-tiles)
    v16bf bf0, bf1;
    {
      const bf16_t* bb0 = &curB[(wn * 32 + fm) * LDS_STR + k0b];
      const bf16_t* bb1 = &curB[(wn * 32 + 16 + fm) * LDS_STR + k0b];
      v8bf l0 = *(const v8bf*)(bb0), h0 = *(const v8bf*)(bb0 + 8);
      v8bf l1 = *(const v8bf*)(bb1), h1 = *(const v8bf*)(bb1 + 8);
      bf0 = __builtin_shufflevector(l0, h0, 0,1,2,3,4,5,6,7,8,9,10,11,12,13,14,15);
      bf1 = __builtin_shufflevector(l1, h1, 0,1,2,3,4,5,6,7,8,9,10,11,12,13,14,15);
    }
#pragma unroll
    for (int mt = 0; mt < 4; ++mt) {
      const bf16_t* ab = &curA[(wm * 64 + mt * 16 + fm) * LDS_STR];
      v8bf alo = *(const v8bf*)(ab + k0a);
      v8bf ahi = *(const v8bf*)(ab + k0a + 8);
      v16bf af = __builtin_shufflevector(alo, ahi,
                   0,1,2,3,4,5,6,7,8,9,10,11,12,13,14,15);
      acc[mt][0] = __builtin_amdgcn_wmma_f32_16x16x32_bf16(
          false, af, false, bf0, (short)0, acc[mt][0], false, false);
      acc[mt][1] = __builtin_amdgcn_wmma_f32_16x16x32_bf16(
          false, af, false, bf1, (short)0, acc[mt][1], false, false);
    }
  }

  // epilogue: C fragment layout -> VGPR r holds row r (+8 for hi half-wave)
  const int hiM  = (lane >> 4) * 8;
  const int ncol = lane & 15;
#pragma unroll
  for (int mt = 0; mt < 4; ++mt)
#pragma unroll
    for (int nt = 0; nt < 2; ++nt) {
      int gn = n0 + wn * 32 + nt * 16 + ncol;
      if (gn >= g.N) continue;
      float bia = g.bias ? g.bias[gn] : 0.f;
      int gmb = m0 + wm * 64 + mt * 16 + hiM;
#pragma unroll
      for (int r = 0; r < 8; ++r) {
        int gm = gmb + r;
        float v = acc[mt][nt][r] * g.alpha + bia;
        if (g.flags & 1) v = fmaxf(v, 0.f);
        long off = (g.flags & 2) ? ((long)gn * g.ldc + gm)
                                 : ((long)gm * g.ldc + gn);
        off += g.sC * (long)z;
        if (g.flags & 4) ((bf16_t*)g.C)[off] = (bf16_t)v;
        else             ((float*)g.C)[off]  = v;
      }
    }
}

// ---------------- sinusoidal RoPE tables: sin/cos[s][i], i=0..31 ------------
__global__ void pos_table_kernel(float* sin_t, float* cos_t) {
  int s = blockIdx.x, i = threadIdx.x;                 // 384 x 32
  float freq = __powf(10000.f, -(float)i / 32.f);
  float a = (float)s * freq;
  sin_t[s * 32 + i] = __sinf(a);
  cos_t[s * 32 + i] = __cosf(a);
}

// ---------------- casts -----------------------------------------------------
__global__ void cast_f32_bf16_kernel(const float* __restrict__ src,
                                     bf16_t* __restrict__ dst, long n) {
  for (long i = (long)blockIdx.x * blockDim.x + threadIdx.x; i < n;
       i += (long)gridDim.x * blockDim.x)
    dst[i] = (bf16_t)src[i];
}

__global__ void cast2d_f32_bf16_kernel(const float* __restrict__ src,
                                       bf16_t* __restrict__ dst,
                                       long rows, int cols, int ldd) {
  long n = rows * cols;
  for (long i = (long)blockIdx.x * blockDim.x + threadIdx.x; i < n;
       i += (long)gridDim.x * blockDim.x) {
    long r = i / cols; int c = (int)(i - r * cols);
    dst[r * ldd + c] = (bf16_t)src[i];
  }
}

// ---------------- mask + RoPE, writes head-major bf16 [B,nH,S,DK] -----------
__global__ void rope_mask_kernel(const float* __restrict__ lin,
                                 const float* __restrict__ mask,
                                 const float* __restrict__ sin_t,
                                 const float* __restrict__ cos_t,
                                 bf16_t* __restrict__ out, int nH, long total) {
  for (long idx = (long)blockIdx.x * blockDim.x + threadIdx.x; idx < total;
       idx += (long)gridDim.x * blockDim.x) {
    long t = idx;
    int d2 = (int)(t & 31); t >>= 5;
    int h  = (int)(t % nH); t /= nH;
    int s  = (int)(t % SS);
    int b  = (int)(t / SS);
    float mk = mask[(long)b * SS + s];
    const float* src = lin + ((long)(b * SS + s) * nH + h) * DKK + 2 * d2;
    float x0 = src[0] * mk, x1 = src[1] * mk;
    float sn = sin_t[s * 32 + d2], cs = cos_t[s * 32 + d2];
    bf16_t* dstp = out + ((long)(b * nH + h) * SS + s) * DKK + 2 * d2;
    dstp[0] = (bf16_t)(x0 * cs - x1 * sn);
    dstp[1] = (bf16_t)(x1 * cs + x0 * sn);
  }
}

// ---------------- symmetric att -> masked softmax -> bf16 adjacency ---------
__global__ __launch_bounds__(256) void adj_softmax_kernel(
    const float* __restrict__ score, bf16_t* __restrict__ adj) {
  const int i = blockIdx.x;
  const int z = blockIdx.y;                 // b*9+h
  const long base = (long)z * SS * SS;
  __shared__ float red[256];
  const int tid = threadIdx.x;
  const bool ok1 = (tid + 256) < SS;
  float v0, v1;
  {
    int j = tid;
    int lo = (i < j) ? i : j, hi = (i < j) ? j : i;
    v0 = score[base + (long)lo * SS + hi];
  }
  if (ok1) {
    int j = tid + 256;
    int lo = (i < j) ? i : j, hi = (i < j) ? j : i;
    v1 = score[base + (long)lo * SS + hi];
  } else v1 = -3.0e38f;
  float m = fmaxf(v0, v1);
  red[tid] = m; __syncthreads();
  for (int s = 128; s > 0; s >>= 1) {
    if (tid < s) red[tid] = fmaxf(red[tid], red[tid + s]);
    __syncthreads();
  }
  m = red[0]; __syncthreads();
  float e0 = __expf(v0 - m);
  float e1 = ok1 ? __expf(v1 - m) : 0.f;
  red[tid] = e0 + e1; __syncthreads();
  for (int s = 128; s > 0; s >>= 1) {
    if (tid < s) red[tid] += red[tid + s];
    __syncthreads();
  }
  float inv = 1.f / red[0];
  adj[base + (long)i * SS + tid] = (bf16_t)((v0 != 0.f) ? e0 * inv : 0.f);
  if (ok1)
    adj[base + (long)i * SS + tid + 256] =
        (bf16_t)((v1 != 0.f) ? e1 * inv : 0.f);
}

// ---------------- sum over 9 relations + ReLU + /9 + LayerNorm -> bf16 ------
__global__ __launch_bounds__(256) void rgat_reduce_ln_kernel(
    const bf16_t* __restrict__ agg, bf16_t* __restrict__ dst, int ldd) {
  const int s = blockIdx.x, b = blockIdx.y;
  const int tid = threadIdx.x;
  __shared__ float red[256];
  float v[3];
#pragma unroll
  for (int t = 0; t < 3; ++t) {
    int c = tid + t * 256;
    float sum = 0.f;
#pragma unroll
    for (int r = 0; r < 9; ++r)
      sum += (float)agg[((long)(b * 9 + r) * SS + s) * HH + c];
    v[t] = fmaxf(sum, 0.f) * (1.0f / 9.0f);
  }
  red[tid] = v[0] + v[1] + v[2]; __syncthreads();
  for (int st = 128; st > 0; st >>= 1) {
    if (tid < st) red[tid] += red[tid + st];
    __syncthreads();
  }
  float mean = red[0] * (1.0f / HH); __syncthreads();
  float var = 0.f;
#pragma unroll
  for (int t = 0; t < 3; ++t) { float d = v[t] - mean; var += d * d; }
  red[tid] = var; __syncthreads();
  for (int st = 128; st > 0; st >>= 1) {
    if (tid < st) red[tid] += red[tid + st];
    __syncthreads();
  }
  float rstd = rsqrtf(red[0] * (1.0f / HH) + 1e-5f);
#pragma unroll
  for (int t = 0; t < 3; ++t)
    dst[(long)(b * SS + s) * ldd + tid + t * 256] =
        (bf16_t)((v[t] - mean) * rstd);
}

// ---------------- upper-triangle gather into [B, P, 16] ---------------------
__global__ void shaking_kernel(const float* __restrict__ scp,
                               const float* __restrict__ thrp,
                               const float* __restrict__ scg,
                               const float* __restrict__ thrg,
                               float* __restrict__ out) {
  const int i = blockIdx.x, b = blockIdx.y;
  const long rowbase = (long)i * SS - ((long)i * (i - 1)) / 2;
  for (int j = i + threadIdx.x; j < SS; j += blockDim.x) {
    long p = rowbase + (j - i);
    float* o = out + ((long)b * PP + p) * 16;
    long ij = (long)i * SS + j;
#pragma unroll
    for (int h = 0; h < 6; ++h)
      o[h] = scp[(long)(b * 6 + h) * SS * SS + ij];
    o[6] = fmaxf(thrp[(long)b * SS * SS + ij], 0.f);
#pragma unroll
    for (int h = 0; h < 8; ++h)
      o[7 + h] = scg[(long)(b * 9 + h) * SS * SS + ij];
    o[15] = fmaxf(thrg[(long)b * SS * SS + ij], 0.f);
  }
}

// ===========================================================================
extern "C" void kernel_launch(void* const* d_in, const int* in_sizes, int n_in,
                              void* d_out, int out_size, void* d_ws,
                              size_t ws_size, hipStream_t stream) {
  (void)in_sizes; (void)n_in; (void)out_size; (void)ws_size;
  const float* x    = (const float*)d_in[0];
  const float* mask = (const float*)d_in[1];

  char* base = (char*)d_ws;
  size_t off = 0;
  auto alloc = [&](size_t bytes) -> void* {
    void* p = base + off;
    off = (off + bytes + 255) & ~(size_t)255;
    return p;
  };
  auto castW = [&](int idx, long n) -> bf16_t* {
    bf16_t* dst = (bf16_t*)alloc((size_t)n * sizeof(bf16_t));
    int blocks = (int)((n + 2047) / 2048);
    if (blocks < 1) blocks = 1;
    if (blocks > 4096) blocks = 4096;
    cast_f32_bf16_kernel<<<blocks, 256, 0, stream>>>((const float*)d_in[idx],
                                                     dst, n);
    return dst;
  };
  auto gemm = [&](const bf16_t* A, long sA, int aDiv, int lda,
                  const bf16_t* B, long sB, int bMod, int ldb,
                  const float* bias, void* C, long sC, int ldc,
                  int M, int N, int K, int nz, float alpha, int flags) {
    GemmArgs g;
    g.A = A; g.B = B; g.bias = bias; g.C = C;
    g.sA = sA; g.sB = sB; g.sC = sC;
    g.aDiv = aDiv; g.bMod = bMod;
    g.M = M; g.N = N; g.K = K; g.lda = lda; g.ldb = ldb; g.ldc = ldc;
    g.alpha = alpha; g.flags = flags;
    dim3 grid((N + TBN - 1) / TBN, M / TBM, nz);
    gemm_bf16_wmma<<<grid, 256, 0, stream>>>(g);
  };
  const int BIG = 1 << 30;
  const float inv = 0.125f;   // 1/sqrt(DK)
  const long MR = (long)BB * SS;   // 6144

  // ---------- persistent allocations ----------
  float* sin_t = (float*)alloc(SS * 32 * 4);
  float* cos_t = (float*)alloc(SS * 32 * 4);
  pos_table_kernel<<<SS, 32, 0, stream>>>(sin_t, cos_t);

  bf16_t* w_q4gt   = castW(2,  64L * 768);
  bf16_t* w_k4gt   = castW(4,  64L * 768);
  bf16_t* w_qg_rep = castW(6,  672L * 768);
  bf16_t* w_qg_sc  = castW(8,  576L * 672);
  bf16_t* w_kg_rep = castW(10, 672L * 768);
  bf16_t* w_kg_sc  = castW(12, 576L * 672);
  bf16_t* w_rgat   = castW(14, 2L * 9 * 768 * 768);
  bf16_t* w_q4t    = castW(15, 64L * 1536);
  bf16_t* w_k4t    = castW(17, 64L * 1536);
  bf16_t* w_qp_rep = castW(19, 960L * 1536);
  bf16_t* w_qp_sc  = castW(21, 384L * 960);
  bf16_t* w_kp_rep = castW(23, 960L * 1536);
  bf16_t* w_kp_sc  = castW(25, 384L * 960);

  bf16_t* inp_bf  = (bf16_t*)alloc((size_t)MR * 1536 * 2);  // [x_bf | feat_bf]
  cast2d_f32_bf16_kernel<<<4096, 256, 0, stream>>>(x, inp_bf, MR, HH, 1536);
  bf16_t* feat_bf = (bf16_t*)alloc((size_t)MR * HH * 2);
  float* scr_tg = (float*)alloc((size_t)BB * SS * SS * 4);
  float* scr_tp = (float*)alloc((size_t)BB * SS * SS * 4);
  float* score  = (float*)alloc((size_t)BB * 9 * SS * SS * 4);

  const size_t mark = off;   // phase-temp reset point

  // ================= phase A: graph-layer Q/K + scores =====================
  float*  tsq_lin = (float*)alloc((size_t)MR * 64 * 4);
  float*  tsk_lin = (float*)alloc((size_t)MR * 64 * 4);
  bf16_t* tsq_bf  = (bf16_t*)alloc((size_t)BB * SS * 64 * 2);
  bf16_t* tsk_bf  = (bf16_t*)alloc((size_t)BB * SS * 64 * 2);
  gemm(inp_bf, 0, 1, 1536, w_q4gt, 0, BIG, 768, (const float*)d_in[3],
       tsq_lin, 0, 64, (int)MR, 64, 768, 1, 1.f, 0);
  gemm(inp_bf, 0, 1, 1536, w_k4gt, 0, BIG, 768, (const float*)d_in[5],
       tsk_lin, 0, 64, (int)MR, 64, 768, 1, 1.f, 0);
  rope_mask_kernel<<<2048, 256, 0, stream>>>(tsq_lin, mask, sin_t, cos_t,
                                             tsq_bf, 1, (long)BB * SS * 32);
  rope_mask_kernel<<<2048, 256, 0, stream>>>(tsk_lin, mask, sin_t, cos_t,
                                             tsk_bf, 1, (long)BB * SS * 32);
  gemm(tsq_bf, (long)SS * 64, 1, 64, tsk_bf, (long)SS * 64, BIG, 64, nullptr,
       scr_tg, (long)SS * SS, SS, SS, SS, 64, BB, inv, 0);

  bf16_t* q_rep = (bf16_t*)alloc((size_t)MR * 672 * 2);
  bf16_t* k_rep = (bf16_t*)alloc((size_t)MR * 672 * 2);
  float*  q_sc  = (float*)alloc((size_t)MR * 576 * 4);
  float*  k_sc  = (float*)alloc((size_t)MR * 576 * 4);
  bf16_t* q_bf  = (bf16_t*)alloc((size_t)BB * 9 * SS * 64 * 2);
  bf16_t* k_bf  = (bf16_t*)alloc((size_t)BB * 9 * SS * 64 * 2);
  gemm(inp_bf, 0, 1, 1536, w_qg_rep, 0, BIG, 768, (const float*)d_in[7],
       q_rep, 0, 672, (int)MR, 672, 768, 1, 1.f, 1 | 4);
  gemm(q_rep, 0, 1, 672, w_qg_sc, 0, BIG, 672, (const float*)d_in[9],
       q_sc, 0, 576, (int)MR, 576, 672, 1, 1.f, 0);
  gemm(inp_bf, 0, 1, 1536, w_kg_rep, 0, BIG, 768, (const float*)d_in[11],
       k_rep, 0, 672, (int)MR, 672, 768, 1, 1.f, 1 | 4);
  gemm(k_rep, 0, 1, 672, w_kg_sc, 0, BIG, 672, (const float*)d_in[13],
       k_sc, 0, 576, (int)MR, 576, 672, 1, 1.f, 0);
  rope_mask_kernel<<<4096, 256, 0, stream>>>(q_sc, mask, sin_t, cos_t,
                                             q_bf, 9, (long)BB * SS * 9 * 32);
  rope_mask_kernel<<<4096, 256, 0, stream>>>(k_sc, mask, sin_t, cos_t,
                                             k_bf, 9, (long)BB * SS * 9 * 32);
  gemm(q_bf, (long)SS * 64, 1, 64, k_bf, (long)SS * 64, BIG, 64, nullptr,
       score, (long)SS * SS, SS, SS, SS, 64, BB * 9, inv, 0);
  off = mark;

  // ================= phase B: adjacency + 2x RGAT ==========================
  bf16_t* adj_bf = (bf16_t*)alloc((size_t)BB * 9 * SS * SS * 2);
  bf16_t* hrT    = (bf16_t*)alloc((size_t)BB * 9 * HH * SS * 2);
  bf16_t* agg    = (bf16_t*)alloc((size_t)BB * 9 * SS * HH * 2);
  adj_softmax_kernel<<<dim3(SS, BB * 9), 256, 0, stream>>>(score, adj_bf);
  const bf16_t* featA = inp_bf;
  int ldaF = 1536;
  for (int l = 0; l < 2; ++l) {
    // hr^T[z=b*9+r] = (feat[b] @ W[l,r]^T)^T  -> stored [H,S] bf16
    gemm(featA, (long)SS * ldaF, 9, ldaF,
         w_rgat + (long)l * 9 * HH * HH, (long)HH * HH, 9, HH,
         nullptr, hrT, (long)HH * SS, SS, SS, HH, HH, BB * 9, 1.f, 2 | 4);
    // agg[z] = relu(adj[z] @ hr[z])  (B operand = hr^T, [N,K] form)
    gemm(adj_bf, (long)SS * SS, 1, SS, hrT, (long)HH * SS, BIG, SS,
         nullptr, agg, (long)SS * HH, HH, SS, HH, SS, BB * 9, 1.f, 1 | 4);
    if (l == 0) {
      rgat_reduce_ln_kernel<<<dim3(SS, BB), 256, 0, stream>>>(agg, feat_bf, HH);
      featA = feat_bf; ldaF = HH;
    } else {
      rgat_reduce_ln_kernel<<<dim3(SS, BB), 256, 0, stream>>>(agg,
                                                              inp_bf + HH, 1536);
    }
  }
  off = mark;

  // ================= phase C: prediction head ==============================
  float*  tq_lin = (float*)alloc((size_t)MR * 64 * 4);
  float*  tk_lin = (float*)alloc((size_t)MR * 64 * 4);
  bf16_t* tq_bf  = (bf16_t*)alloc((size_t)BB * SS * 64 * 2);
  bf16_t* tk_bf  = (bf16_t*)alloc((size_t)BB * SS * 64 * 2);
  gemm(inp_bf, 0, 1, 1536, w_q4t, 0, BIG, 1536, (const float*)d_in[16],
       tq_lin, 0, 64, (int)MR, 64, 1536, 1, 1.f, 0);
  gemm(inp_bf, 0, 1, 1536, w_k4t, 0, BIG, 1536, (const float*)d_in[18],
       tk_lin, 0, 64, (int)MR, 64, 1536, 1, 1.f, 0);
  rope_mask_kernel<<<2048, 256, 0, stream>>>(tq_lin, mask, sin_t, cos_t,
                                             tq_bf, 1, (long)BB * SS * 32);
  rope_mask_kernel<<<2048, 256, 0, stream>>>(tk_lin, mask, sin_t, cos_t,
                                             tk_bf, 1, (long)BB * SS * 32);
  gemm(tq_bf, (long)SS * 64, 1, 64, tk_bf, (long)SS * 64, BIG, 64, nullptr,
       scr_tp, (long)SS * SS, SS, SS, SS, 64, BB, inv, 0);

  bf16_t* qp_rep = (bf16_t*)alloc((size_t)MR * 960 * 2);
  bf16_t* kp_rep = (bf16_t*)alloc((size_t)MR * 960 * 2);
  float*  qp_sc  = (float*)alloc((size_t)MR * 384 * 4);
  float*  kp_sc  = (float*)alloc((size_t)MR * 384 * 4);
  bf16_t* qp_bf  = (bf16_t*)alloc((size_t)BB * 6 * SS * 64 * 2);
  bf16_t* kp_bf  = (bf16_t*)alloc((size_t)BB * 6 * SS * 64 * 2);
  float*  scp    = (float*)alloc((size_t)BB * 6 * SS * SS * 4);
  gemm(inp_bf, 0, 1, 1536, w_qp_rep, 0, BIG, 1536, (const float*)d_in[20],
       qp_rep, 0, 960, (int)MR, 960, 1536, 1, 1.f, 1 | 4);
  gemm(qp_rep, 0, 1, 960, w_qp_sc, 0, BIG, 960, (const float*)d_in[22],
       qp_sc, 0, 384, (int)MR, 384, 960, 1, 1.f, 0);
  gemm(inp_bf, 0, 1, 1536, w_kp_rep, 0, BIG, 1536, (const float*)d_in[24],
       kp_rep, 0, 960, (int)MR, 960, 1536, 1, 1.f, 1 | 4);
  gemm(kp_rep, 0, 1, 960, w_kp_sc, 0, BIG, 960, (const float*)d_in[26],
       kp_sc, 0, 384, (int)MR, 384, 960, 1, 1.f, 0);
  rope_mask_kernel<<<4096, 256, 0, stream>>>(qp_sc, mask, sin_t, cos_t,
                                             qp_bf, 6, (long)BB * SS * 6 * 32);
  rope_mask_kernel<<<4096, 256, 0, stream>>>(kp_sc, mask, sin_t, cos_t,
                                             kp_bf, 6, (long)BB * SS * 6 * 32);
  gemm(qp_bf, (long)SS * 64, 1, 64, kp_bf, (long)SS * 64, BIG, 64, nullptr,
       scp, (long)SS * SS, SS, SS, SS, 64, BB * 6, inv, 0);

  // ================= final gather ==========================================
  shaking_kernel<<<dim3(SS, BB), 256, 0, stream>>>(scp, scr_tp, score, scr_tg,
                                                   (float*)d_out);
}
